// ModalDecoder_850403525409
// MI455X (gfx1250) — matrix operand order repaired and checked by reading, not compiled, not run on hardware
//
#include <hip/hip_runtime.h>

// ---------------------------------------------------------------------------
// ModalDecoder: out[b,n,f,o] = sum_d (x[b,n,d] + bias[idx[b,f],d]) * W[idx[b,f],d,o]
//   =  (x[b] @ W[v])[n,o]  +  c[b,f,o],   c = bias[v] @ W[v]
// B=4 N=1024 D=512 F=16 O=64 V=64.
// GEMM via bf16x3 split on v_wmma_f32_16x16x32_bf16.  A and B pre-split into
// exact WMMA operand layouts in d_ws.  B flows through a 4-buffer LDS pipeline
// (depth-2 prefetch) filled by global_load_async_to_lds_b128 with threshold
// s_wait_asynccnt; raw split-barrier (no vmem fence) separates the stages so
// A global loads are never drained at the barrier.
// ---------------------------------------------------------------------------

typedef __attribute__((ext_vector_type(16))) __bf16 v16bf;
typedef __attribute__((ext_vector_type(8)))  float  v8f;
typedef int v4i __attribute__((vector_size(16)));   // matches builtin param type

#define B_DIM 4
#define N_DIM 1024
#define D_DIM 512
#define F_DIM 16
#define O_DIM 64
#define V_DIM 64
#define KC_N  (D_DIM / 32)                  // 16 K-chunks of 32
#define OT_N  (O_DIM / 16)                  // 4 o-tiles of 16
// packed B plane: [v][kc][t][lane(32)][j(16)] bf16 -> per v: 16*4*512 = 32768
#define WPACK_PER_V (KC_N * OT_N * 32 * 16)
#define WPACK_TOTAL ((size_t)V_DIM * WPACK_PER_V)
// packed A plane: [b][nt16(64)][kc(16)][lane(32)][j(16)] bf16
#define XPACK_TOTAL ((size_t)B_DIM * N_DIM * D_DIM)

#define AS1 __attribute__((address_space(1)))
#define AS3 __attribute__((address_space(3)))

#if __has_builtin(__builtin_amdgcn_global_load_async_to_lds_b128)
#define HAVE_ASYNC_LDS 1
#endif

__device__ __forceinline__ void stage_b128(const uint4* g, uint4* l) {
#ifdef HAVE_ASYNC_LDS
    __builtin_amdgcn_global_load_async_to_lds_b128((AS1 v4i*)g, (AS3 v4i*)l, 0, 0);
#else
    *l = *g;
#endif
}

#ifdef HAVE_ASYNC_LDS
#if __has_builtin(__builtin_amdgcn_s_wait_asynccnt)
#define STAGE_WAIT(n) __builtin_amdgcn_s_wait_asynccnt(n)
#else
#define STAGE_WAIT(n) asm volatile("s_wait_asynccnt " #n ::: "memory")
#endif
#else
#define STAGE_WAIT(n) ((void)0)
#endif

// Workgroup split-barrier without the __syncthreads vmem fence.  Ordering of
// the async LDS fills is provided by STAGE_WAIT before the signal; LDS reads
// are all consumed by WMMAs pre-barrier (dscnt==0), with a free s_wait_dscnt
// as insurance.  asm memory fences stop compiler-level motion across it.
__device__ __forceinline__ void wg_barrier() {
#if __has_builtin(__builtin_amdgcn_s_barrier_signal) && \
    __has_builtin(__builtin_amdgcn_s_barrier_wait)
    asm volatile("" ::: "memory");
    asm volatile("s_wait_dscnt 0x0" ::: "memory");   // no-op when already 0
    __builtin_amdgcn_s_barrier_signal(-1);
    __builtin_amdgcn_s_barrier_wait(-1);
    asm volatile("" ::: "memory");
#else
    __syncthreads();
#endif
}

// ---------------------------------------------------------------------------
// Kernel 1: repack W[v,d,o] (fp32 row-major) into WMMA B-operand layout, split
// into bf16 hi/lo planes.  B operand 32x16 (KxN), wave32:
//   lane L: column N = L%16, half = L/16, element j (0..15) -> K = half*16 + j (order chosen here, same as A; not AMD's table order)
// ---------------------------------------------------------------------------
__global__ __launch_bounds__(256) void pack_w_kernel(const float* __restrict__ wt,
                                                     __bf16* __restrict__ whi,
                                                     __bf16* __restrict__ wlo) {
    const int tid  = blockIdx.x * 256 + threadIdx.x;   // 0 .. 131071
    const int vv   = tid >> 11;
    const int rem  = tid & 2047;
    const int kc   = rem >> 7;
    const int t    = (rem >> 5) & 3;
    const int lane = rem & 31;
    const int half = lane >> 4;
    const int l16  = lane & 15;

    const int o     = t * 16 + l16;
    const int dbase = kc * 32 + half * 16;
    const float* src = wt + (size_t)vv * (D_DIM * O_DIM) + (size_t)dbase * O_DIM + o;
    const size_t ob  = (size_t)tid * 16;

#pragma unroll
    for (int j = 0; j < 16; ++j) {
        float f  = src[(size_t)j * O_DIM];
        __bf16 h = (__bf16)f;
        whi[ob + j] = h;
        wlo[ob + j] = (__bf16)(f - (float)h);
    }
}

// ---------------------------------------------------------------------------
// Kernel 2: repack x[b,n,d] (fp32) into WMMA A-operand layout, bf16 hi/lo.
// A operand 16x32 (MxK), wave32: lane L -> row M = L%16, half = L/16,
//   element j (0..15) -> K = half*16 + j (order chosen here, same as B; not AMD's table order)
// ---------------------------------------------------------------------------
__global__ __launch_bounds__(256) void pack_x_kernel(const float* __restrict__ x,
                                                     __bf16* __restrict__ xhi,
                                                     __bf16* __restrict__ xlo) {
    const int tid  = blockIdx.x * 256 + threadIdx.x;   // 0 .. 131071
    const int b    = tid >> 15;                        // 64*16*32 per b
    const int rem  = tid & 32767;
    const int nt16 = rem >> 9;                         // 16*32 per n-tile
    const int kc   = (rem >> 5) & 15;
    const int lane = rem & 31;
    const int half = lane >> 4;
    const int l16  = lane & 15;

    const float* xrow = x + ((size_t)b * N_DIM + nt16 * 16 + l16) * D_DIM;
    const int k0 = kc * 32 + half * 16;
    const float4* pa = (const float4*)(xrow + k0);
    const float4* pb = (const float4*)(xrow + k0 + 8);
    float4 a0 = pa[0], a1 = pa[1];
    float4 a2 = pb[0], a3 = pb[1];
    float av[16] = {a0.x, a0.y, a0.z, a0.w, a1.x, a1.y, a1.z, a1.w,
                    a2.x, a2.y, a2.z, a2.w, a3.x, a3.y, a3.z, a3.w};

    const size_t ob = (size_t)tid * 16;
#pragma unroll
    for (int j = 0; j < 16; ++j) {
        __bf16 h = (__bf16)av[j];
        xhi[ob + j] = h;
        xlo[ob + j] = (__bf16)(av[j] - (float)h);
    }
}

// ---------------------------------------------------------------------------
// Kernel 3: c[b,f,o] = sum_d bias[v,d] * W[v,d,o]  (fp32, tiny)
// ---------------------------------------------------------------------------
__global__ __launch_bounds__(256) void cbias_kernel(const int* __restrict__ idx,
                                                    const float* __restrict__ bias,
                                                    const float* __restrict__ wt,
                                                    float* __restrict__ cb) {
    const int tid = blockIdx.x * 256 + threadIdx.x;    // 0 .. 4095
    const int bf  = tid >> 6;
    const int o   = tid & 63;
    const int v   = idx[bf];
    const float* bp = bias + (size_t)v * D_DIM;
    const float* wp = wt + (size_t)v * (D_DIM * O_DIM) + o;
    float s = 0.0f;
    for (int d = 0; d < D_DIM; ++d)
        s = fmaf(bp[d], wp[(size_t)d * O_DIM], s);
    cb[tid] = s;
}

// ---------------------------------------------------------------------------
// Kernel 4: main GEMM.  grid = (N/128, F, B), 256 threads = 8 waves.
// Wave w owns a 16x64 strip (4 o-tile v8f accumulators).
// B: 4-buffer LDS pipeline (4 KB/buffer/plane), chunk kc+2 staged at iter kc
//    via async-to-LDS; s_wait_asynccnt 4 retires exactly chunk kc (in-order).
//    Target buffer (kc+2)&3 is disjoint from {kc-1,kc,kc+1}&3, so pre-barrier
//    issue cannot race laggard readers.
// A: pre-packed bf16 hi/lo, software-pipelined one chunk ahead; with the raw
//    split-barrier there is no loadcnt drain, so A latency hides under WMMAs.
// ---------------------------------------------------------------------------
__global__ __launch_bounds__(256) void modal_main_kernel(
        const __bf16* __restrict__ xhi,
        const __bf16* __restrict__ xlo,
        const int*    __restrict__ idx,
        const __bf16* __restrict__ whi,
        const __bf16* __restrict__ wlo,
        const float*  __restrict__ cb,
        float*        __restrict__ out) {
    __shared__ __align__(16) __bf16 sHi[4 * OT_N * 512];   // 4 x 4 KB
    __shared__ __align__(16) __bf16 sLo[4 * OT_N * 512];

    const int nt   = blockIdx.x;
    const int f    = blockIdx.y;
    const int b    = blockIdx.z;
    const int tid  = threadIdx.x;
    const int wave = tid >> 5;
    const int lane = tid & 31;
    const int half = lane >> 4;
    const int l16  = lane & 15;

    const int v = idx[b * F_DIM + f];

    // pre-packed A: block (b, nt*8+wave, kc), lane-contiguous 32B
    const size_t abase = (((size_t)b * 64 + nt * 8 + wave) * KC_N) * 512 + lane * 16;
    const __bf16* aHi = xhi + abase;
    const __bf16* aLo = xlo + abase;

    const uint4* gHi = (const uint4*)(whi + (size_t)v * WPACK_PER_V); // 256 uint4 per kc
    const uint4* gLo = (const uint4*)(wlo + (size_t)v * WPACK_PER_V);
    uint4* lHi = (uint4*)sHi;                                         // 256 uint4 per buf
    uint4* lLo = (uint4*)sLo;

    v8f acc[OT_N];
#pragma unroll
    for (int t = 0; t < OT_N; ++t) {
        v8f z = {0.f, 0.f, 0.f, 0.f, 0.f, 0.f, 0.f, 0.f};
        acc[t] = z;
    }

    // prologue: stage chunks 0,1 into buffers 0,1; load A chunk 0
    stage_b128(&gHi[tid],       &lHi[tid]);
    stage_b128(&gLo[tid],       &lLo[tid]);
    stage_b128(&gHi[256 + tid], &lHi[256 + tid]);
    stage_b128(&gLo[256 + tid], &lLo[256 + tid]);
    v16bf ahi = *(const v16bf*)(aHi);
    v16bf alo = *(const v16bf*)(aLo);

    for (int kc = 0; kc < KC_N; ++kc) {
        const int buf = kc & 3;

        if (kc + 2 < KC_N) {   // depth-2 prefetch into buffer (kc+2)&3
            const int pb = (kc + 2) & 3;
            stage_b128(&gHi[(kc + 2) * 256 + tid], &lHi[pb * 256 + tid]);
            stage_b128(&gLo[(kc + 2) * 256 + tid], &lLo[pb * 256 + tid]);
        }
        // retire exactly chunk kc (async completes in order); literal immediates
        if      (kc < KC_N - 2) STAGE_WAIT(4);
        else if (kc < KC_N - 1) STAGE_WAIT(2);
        else                    STAGE_WAIT(0);
        wg_barrier();          // all threads' chunk-kc LDS writes visible

        // A pipeline: fetch next chunk's A; latency hides under the WMMAs
        v16bf ahi_n, alo_n;
        if (kc + 1 < KC_N) {
            ahi_n = *(const v16bf*)(aHi + (kc + 1) * 512);
            alo_n = *(const v16bf*)(aLo + (kc + 1) * 512);
        }

        const __bf16* bH = sHi + buf * (OT_N * 512);
        const __bf16* bL = sLo + buf * (OT_N * 512);
#pragma unroll
        for (int t = 0; t < OT_N; ++t) {
            v16bf bh = *(const v16bf*)(bH + t * 512 + lane * 16);
            v16bf bl = *(const v16bf*)(bL + t * 512 + lane * 16);
            // bf16x3: hi*hi + lo*hi + hi*lo, fp32 accumulate
            acc[t] = __builtin_amdgcn_wmma_f32_16x16x32_bf16(
                         false, ahi, false, bh, (short)0, acc[t], false, false);
            acc[t] = __builtin_amdgcn_wmma_f32_16x16x32_bf16(
                         false, alo, false, bh, (short)0, acc[t], false, false);
            acc[t] = __builtin_amdgcn_wmma_f32_16x16x32_bf16(
                         false, ahi, false, bl, (short)0, acc[t], false, false);
        }
        ahi = ahi_n;
        alo = alo_n;
    }

    // epilogue: add bias-constant, store fp32.
    // C/D layout: VGPR r, lane L -> M = r + 8*(L/16), N = L%16
    const float* cbp = cb + (size_t)(b * F_DIM + f) * O_DIM;
#pragma unroll
    for (int t = 0; t < OT_N; ++t) {
        const float cbv = cbp[t * 16 + l16];
        const int   o   = t * 16 + l16;
#pragma unroll
        for (int r = 0; r < 8; ++r) {
            const int n = nt * 128 + wave * 16 + half * 8 + r;
            out[(((size_t)b * N_DIM + n) * F_DIM + f) * O_DIM + o] = acc[t][r] + cbv;
        }
    }
}

// ---------------------------------------------------------------------------
extern "C" void kernel_launch(void* const* d_in, const int* in_sizes, int n_in,
                              void* d_out, int out_size, void* d_ws, size_t ws_size,
                              hipStream_t stream) {
    (void)in_sizes; (void)n_in; (void)out_size; (void)ws_size;

    const float* x    = (const float*)d_in[0];   // [4,1024,512] f32
    const int*   idx  = (const int*)  d_in[1];   // [4,16] i32
    const float* bias = (const float*)d_in[2];   // [64,512] f32
    const float* wt   = (const float*)d_in[3];   // [64, 512*64] f32
    float*       out  = (float*)d_out;           // [4,1024,16,64] f32

    __bf16* whi = (__bf16*)d_ws;                 // 4 MB
    __bf16* wlo = whi + WPACK_TOTAL;             // 4 MB
    __bf16* xhi = wlo + WPACK_TOTAL;             // 4 MB
    __bf16* xlo = xhi + XPACK_TOTAL;             // 4 MB
    float*  cbp = (float*)(xlo + XPACK_TOTAL);   // 16 KB

    pack_w_kernel<<<dim3(512), dim3(256), 0, stream>>>(wt, whi, wlo);
    pack_x_kernel<<<dim3(512), dim3(256), 0, stream>>>(x, xhi, xlo);
    cbias_kernel <<<dim3(16),  dim3(256), 0, stream>>>(idx, bias, wt, cbp);

    dim3 grid(N_DIM / 128, F_DIM, B_DIM);        // (8,16,4)
    modal_main_kernel<<<grid, dim3(256), 0, stream>>>(xhi, xlo, idx, whi, wlo, cbp, out);
}
